// AllDeepSet_g_28166395527447
// MI455X (gfx1250) — compile-verified, matrix operand order checked by reading, not run on hardware
//
#include <hip/hip_runtime.h>
#include <hip/hip_bf16.h>

// ---------------------------------------------------------------------------
// AllDeepSet for MI455X (gfx1250): bf16 WMMA GEMMs (f32 accumulate), weights
// staged in LDS per block, fused bias+ReLU, LDS transpose between GEMMs,
// atomic scatter-add message passing.
// ---------------------------------------------------------------------------

#define DD   128      // feature dim
#define NL   2        // layers
#define NG   64       // graphs
#define NC   40       // classes
#define MDIM 20000    // hyperedges (reference constant; not derivable from sizes)

typedef __attribute__((ext_vector_type(16))) __bf16 v16bf;
typedef __attribute__((ext_vector_type(8)))  float  v8f;

// Load one 16x32 bf16 A-fragment from an f32 row (row-major, contiguous cols).
// A layout (CDNA5 16-bit A 16x32): lanes 0-15 hold row m=lane, K = {0..7,16..23};
// lanes 16-31 hold row m=lane-16, K = {8..15,24..31}. koff = (lane>=16)*8.
__device__ inline v16bf load_frag_f32(const float* __restrict__ row, int kb, int koff) {
    v16bf a;
    const float* p0 = row + kb * 32 + koff;
    const float* p1 = p0 + 16;
#pragma unroll
    for (int j = 0; j < 8; ++j) {
        a[j]     = (__bf16)p0[j];
        a[j + 8] = (__bf16)p1[j];
    }
    return a;
}

// Pack a [128x128] f32 weight (row-major, W[k][n]) into WMMA B-fragment layout:
// frag index (t,kb): lane l holds column n = t*16 + (l&15), K pattern mirrors A.
__global__ void pack_weight_kernel(const float* __restrict__ W, __bf16* __restrict__ out) {
    int idx = blockIdx.x * blockDim.x + threadIdx.x;   // 0..16383
    if (idx >= 8 * 4 * 32 * 16) return;
    int j    = idx & 15;
    int lane = (idx >> 4) & 31;
    int kb   = (idx >> 9) & 3;
    int t    = idx >> 11;
    int n    = t * 16 + (lane & 15);
    int koff = (lane >> 4) << 3;
    int k    = kb * 32 + koff + ((j < 8) ? j : (j - 8 + 16));
    out[idx] = (__bf16)W[k * DD + n];
}

// Y = relu( relu(X @ W1 + b1) @ W2 + b2 ), X,Y: [R,128] f32 (in-place safe:
// each wave reads only its own 16-row strip before storing it).
#define MLP_WAVES 8
#define WFRAGS (8 * 4 * 32)          // 1024 v16bf fragments per matrix
__global__ __launch_bounds__(MLP_WAVES * 32)
void mlp2_relu_kernel(const float* __restrict__ X, float* __restrict__ Y,
                      const __bf16* __restrict__ W1p, const float* __restrict__ b1,
                      const __bf16* __restrict__ W2p, const float* __restrict__ b2,
                      int R) {
    // LDS: two packed weight matrices (32 KB each) + per-wave transpose slabs.
    __shared__ __align__(32) __bf16 w1s[WFRAGS * 16];
    __shared__ __align__(32) __bf16 w2s[WFRAGS * 16];
    __shared__ float slab_all[MLP_WAVES][16][132];   // padded: no bank conflicts

    const int lane  = threadIdx.x & 31;
    const int wave  = threadIdx.x >> 5;
    const int rl    = lane & 15;
    const int koff  = (lane >> 4) << 3;
    const int mbase = (lane >> 4) << 3;

    // ---- cooperative weight staging: 64 KB global -> LDS (b128 copies) ----
    {
        int4* d1 = (int4*)w1s;  const int4* s1 = (const int4*)W1p;
        int4* d2 = (int4*)w2s;  const int4* s2 = (const int4*)W2p;
        for (int i = threadIdx.x; i < 2048; i += MLP_WAVES * 32) d1[i] = s1[i];
        for (int i = threadIdx.x; i < 2048; i += MLP_WAVES * 32) d2[i] = s2[i];
    }
    __syncthreads();

    int tile = blockIdx.x * MLP_WAVES + wave;     // wave-uniform
    const int ntiles = (R + 15) >> 4;
    const bool active = (tile < ntiles);
    if (!active) tile = 0;                        // compute on tile 0, discard
    const int row0 = tile * 16;

    const v16bf* Wf1 = (const v16bf*)w1s;
    const v16bf* Wf2 = (const v16bf*)w2s;

    // ---------------- GEMM1: h = relu(X @ W1 + b1) ----------------
    v8f acc[8];
#pragma unroll
    for (int t = 0; t < 8; ++t) {
        float bv = b1[t * 16 + rl];
#pragma unroll
        for (int i = 0; i < 8; ++i) acc[t][i] = bv;
    }
    int rload = row0 + rl; if (rload > R - 1) rload = R - 1;
    const float* xrow = X + (size_t)rload * DD;
#pragma unroll
    for (int kb = 0; kb < 4; ++kb) {
        v16bf a = load_frag_f32(xrow, kb, koff);
#pragma unroll
        for (int t = 0; t < 8; ++t) {
            acc[t] = __builtin_amdgcn_wmma_f32_16x16x32_bf16(
                false, a, false, Wf1[(t * 4 + kb) * 32 + lane],
                (short)0, acc[t], false, false);
        }
    }
    // relu + stage 16x128 f32 tile in LDS (C-layout -> row-major transpose)
    float* slab = &slab_all[wave][0][0];
#pragma unroll
    for (int t = 0; t < 8; ++t) {
#pragma unroll
        for (int v = 0; v < 8; ++v) {
            float h = acc[t][v];
            slab[(mbase + v) * 132 + t * 16 + rl] = (h > 0.f) ? h : 0.f;
        }
    }
    __syncthreads();

    // ---------------- GEMM2: y = relu(h @ W2 + b2) ----------------
    v8f acc2[8];
#pragma unroll
    for (int t = 0; t < 8; ++t) {
        float bv = b2[t * 16 + rl];
#pragma unroll
        for (int i = 0; i < 8; ++i) acc2[t][i] = bv;
    }
    const float* hrow = slab + rl * 132;
#pragma unroll
    for (int kb = 0; kb < 4; ++kb) {
        v16bf a = load_frag_f32(hrow, kb, koff);
#pragma unroll
        for (int t = 0; t < 8; ++t) {
            acc2[t] = __builtin_amdgcn_wmma_f32_16x16x32_bf16(
                false, a, false, Wf2[(t * 4 + kb) * 32 + lane],
                (short)0, acc2[t], false, false);
        }
    }
    if (active) {
#pragma unroll
        for (int t = 0; t < 8; ++t) {
#pragma unroll
            for (int v = 0; v < 8; ++v) {
                int r = row0 + mbase + v;
                if (r < R) {
                    float h = acc2[t][v];
                    Y[(size_t)r * DD + t * 16 + rl] = (h > 0.f) ? h : 0.f;
                }
            }
        }
    }
}

__global__ void zero_kernel(float* __restrict__ p, long long n) {
    long long i = (long long)blockIdx.x * blockDim.x + threadIdx.x;
    if (i < n) p[i] = 0.f;
}

// One wave per incidence edge: out[didx[e]] += feat[sidx[e]]  (128 f32)
__global__ void scatter_add_kernel(const float* __restrict__ feat,
                                   const int* __restrict__ sidx,
                                   const int* __restrict__ didx,
                                   float* __restrict__ out, int E) {
    int e = blockIdx.x * (blockDim.x >> 5) + (threadIdx.x >> 5);
    if (e >= E) return;
    int lane = threadIdx.x & 31;
    const float4 v = ((const float4*)(feat + (size_t)sidx[e] * DD))[lane];
    float* d = out + (size_t)didx[e] * DD + lane * 4;
    atomicAdd(d + 0, v.x); atomicAdd(d + 1, v.y);
    atomicAdd(d + 2, v.z); atomicAdd(d + 3, v.w);
}

// One wave per node: sums[batch[n]] += x[n]; counts[batch[n]] += 1
__global__ void segment_accum_kernel(const float* __restrict__ xN,
                                     const int* __restrict__ batch,
                                     float* __restrict__ sums,
                                     float* __restrict__ counts, int n) {
    int v = blockIdx.x * (blockDim.x >> 5) + (threadIdx.x >> 5);
    if (v >= n) return;
    int lane = threadIdx.x & 31;
    int b = batch[v];
    const float4 val = ((const float4*)(xN + (size_t)v * DD))[lane];
    float* d = sums + (size_t)b * DD + lane * 4;
    atomicAdd(d + 0, val.x); atomicAdd(d + 1, val.y);
    atomicAdd(d + 2, val.z); atomicAdd(d + 3, val.w);
    if (lane == 0) atomicAdd(counts + b, 1.0f);
}

__global__ void readout_div_kernel(float* __restrict__ sums, const float* __restrict__ counts) {
    int i = blockIdx.x * blockDim.x + threadIdx.x;   // NG*DD
    if (i >= NG * DD) return;
    float c = counts[i >> 7];
    sums[i] = sums[i] / (c > 1.f ? c : 1.f);
}

__global__ void cls1_kernel(const float* __restrict__ r, const float* __restrict__ W1,
                            const float* __restrict__ b1, float* __restrict__ h) {
    int i = blockIdx.x * blockDim.x + threadIdx.x;   // NG*DD
    if (i >= NG * DD) return;
    int g = i >> 7, j = i & 127;
    float s = b1[j];
    const float* rr = r + (size_t)g * DD;
    for (int k = 0; k < DD; ++k) s += rr[k] * W1[k * DD + j];
    h[i] = (s > 0.f) ? s : 0.f;
}

__global__ void cls2_kernel(const float* __restrict__ h, const float* __restrict__ W2,
                            const float* __restrict__ b2, float* __restrict__ out) {
    int i = blockIdx.x * blockDim.x + threadIdx.x;   // NG*NC
    if (i >= NG * NC) return;
    int g = i / NC, c = i % NC;
    float s = b2[c];
    const float* hh = h + (size_t)g * DD;
    for (int k = 0; k < DD; ++k) s += hh[k] * W2[k * NC + c];
    out[i] = s;
}

static inline size_t align256(size_t x) { return (x + 255) & ~(size_t)255; }

extern "C" void kernel_launch(void* const* d_in, const int* in_sizes, int n_in,
                              void* d_out, int out_size, void* d_ws, size_t ws_size,
                              hipStream_t stream) {
    const int N = in_sizes[0] / DD;     // 100000
    const int E = in_sizes[1];          // 800000
    const int M = MDIM;                 // 20000

    const float* X        = (const float*)d_in[0];
    const int*   v2e_src  = (const int*)d_in[1];
    const int*   v2e_dst  = (const int*)d_in[2];
    const int*   batch    = (const int*)d_in[3];
    const float* cls_W1   = (const float*)d_in[20];
    const float* cls_b1   = (const float*)d_in[21];
    const float* cls_W2   = (const float*)d_in[22];
    const float* cls_b2   = (const float*)d_in[23];

    // ---- workspace carve-out ----
    char* ws = (char*)d_ws;
    size_t off = 0;
    float* xN = (float*)(ws + off);  off += align256((size_t)N * DD * 4);
    float* xM = (float*)(ws + off);  off += align256((size_t)M * DD * 4);
    float* sums   = (float*)(ws + off);  off += align256((size_t)NG * DD * 4);  // counts directly after (32 KB is 256-aligned)
    float* counts = (float*)(ws + off);  off += align256((size_t)NG * 4);
    float* hcls   = (float*)(ws + off);  off += align256((size_t)NG * DD * 4);
    __bf16* packed = (__bf16*)(ws + off); // 16 matrices x 16384 bf16 = 512 KB
    const size_t FRAG = 8 * 4 * 32 * 16;  // 16384 bf16 per matrix

    // ---- 1) pack all 16 layered weight matrices to bf16 WMMA fragments ----
    for (int role = 0; role < 4; ++role) {
        for (int l = 0; l < NL; ++l) {
            for (int m = 0; m < 2; ++m) {  // m=0 -> W1, m=1 -> W2
                const float* src = (const float*)d_in[4 + role * 4 + m * 2] + (size_t)l * DD * DD;
                __bf16* dst = packed + (((size_t)role * NL + l) * 2 + m) * FRAG;
                pack_weight_kernel<<<64, 256, 0, stream>>>(src, dst);
            }
        }
    }

    // ---- 2) x = X ----
    hipMemcpyAsync(xN, X, (size_t)N * DD * sizeof(float), hipMemcpyDeviceToDevice, stream);

    auto launch_mlp = [&](float* Xb, float* Yb, int R, int role, int l) {
        const __bf16* w1 = packed + (((size_t)role * NL + l) * 2 + 0) * FRAG;
        const __bf16* w2 = packed + (((size_t)role * NL + l) * 2 + 1) * FRAG;
        const float* b1 = (const float*)d_in[5 + role * 4] + (size_t)l * DD;
        const float* b2 = (const float*)d_in[7 + role * 4] + (size_t)l * DD;
        int tiles = (R + 15) / 16;
        int blocks = (tiles + MLP_WAVES - 1) / MLP_WAVES;
        mlp2_relu_kernel<<<blocks, MLP_WAVES * 32, 0, stream>>>(Xb, Yb, w1, b1, w2, b2, R);
    };

    const long long nElemN = (long long)N * DD;
    const long long nElemM = (long long)M * DD;
    const int scatterBlocks = (E + 7) / 8;          // 8 waves/block, 1 wave/edge

    // ---- 3) message-passing layers ----
    for (int l = 0; l < NL; ++l) {
        // V2E: node MLP (in place) -> scatter into edges -> edge MLP
        launch_mlp(xN, xN, N, /*ve_enc*/0, l);
        zero_kernel<<<(int)((nElemM + 255) / 256), 256, 0, stream>>>(xM, nElemM);
        scatter_add_kernel<<<scatterBlocks, 256, 0, stream>>>(xN, v2e_src, v2e_dst, xM, E);
        launch_mlp(xM, xM, M, /*ve_dec*/1, l);
        // E2V: edge MLP (in place) -> scatter into nodes -> node MLP
        launch_mlp(xM, xM, M, /*ev_enc*/2, l);
        zero_kernel<<<(int)((nElemN + 255) / 256), 256, 0, stream>>>(xN, nElemN);
        scatter_add_kernel<<<scatterBlocks, 256, 0, stream>>>(xM, v2e_dst, v2e_src, xN, E);
        launch_mlp(xN, xN, N, /*ev_dec*/3, l);
    }

    // ---- 4) segment-mean readout + classifier ----
    zero_kernel<<<(NG * DD + NG + 255) / 256, 256, 0, stream>>>(sums, NG * DD + NG);
    segment_accum_kernel<<<(N + 7) / 8, 256, 0, stream>>>(xN, batch, sums, counts, N);
    readout_div_kernel<<<(NG * DD + 255) / 256, 256, 0, stream>>>(sums, counts);
    cls1_kernel<<<(NG * DD + 255) / 256, 256, 0, stream>>>(sums, cls_W1, cls_b1, hcls);
    cls2_kernel<<<(NG * NC + 255) / 256, 256, 0, stream>>>(hcls, cls_W2, cls_b2, (float*)d_out);
}
